// Self_Attn_12378095747263
// MI455X (gfx1250) — matrix operand order repaired and checked by reading, not compiled, not run on hardware
//
#include <hip/hip_runtime.h>

// ---------------------------------------------------------------------------
// Self-attention (B=4, C=256, N=4096, dqk=32) for gfx1250 (MI455X, wave32).
// bf16 WMMA (v_wmma_f32_16x16x32_bf16) for all three GEMMs; fp32 accumulate.
// Attention streams (256 MB > 192 MB L2) use non-temporal hints; softmax row
// staging uses the CDNA5 async global->LDS path (ASYNCcnt) when available.
// Outputs (concatenated): final[4,256,64,64], weighted_out[4,256,64,64],
// attention[4,4096,4096].
// ---------------------------------------------------------------------------

typedef __attribute__((ext_vector_type(16))) __bf16 bf16x16;
typedef __attribute__((ext_vector_type(8)))  float  f32x8;
typedef __attribute__((ext_vector_type(4)))  float  f32x4;   // native vec for NT ld/st
typedef __attribute__((ext_vector_type(4)))  int    v4i;

typedef __attribute__((address_space(1))) v4i gv4i;  // global int4 (async src)
typedef __attribute__((address_space(3))) v4i sv4i;  // LDS int4 (async dst)

#define B_   4
#define C_   256
#define DQK_ 32
#define N_   4096

#define HAVE_ASYNC_LDS (__has_builtin(__builtin_amdgcn_global_load_async_to_lds_b128) && \
                        __has_builtin(__builtin_amdgcn_s_wait_asynccnt))

__device__ __forceinline__ f32x8 wmma_bf16(bf16x16 a, bf16x16 b, f32x8 c) {
  // 8 args: (neg_a, A, neg_b, B, c_mod, C, reuse_a, reuse_b)
  return __builtin_amdgcn_wmma_f32_16x16x32_bf16(false, a, false, b,
                                                 (short)0, c, false, false);
}

// A-matrix (16x32 bf16) lane layout: lane = (row r in 0..15, half h).
// element j of the 16-wide vector holds K = 16*h + j.
__device__ __forceinline__ int a_kk(int j, int h) {
  return 16 * h + j;
}

// ---------------------------------------------------------------------------
// Kernel 1: fused QKV projection.  out[o,n] = sum_c W[o,c]*x[b,c,n] + bias[o]
// 320 output rows per batch (32 q + 32 k + 256 v), stored as bf16 in ws.
// One 16x16 tile per wave, K-loop over C=256 in chunks of 32.
// ---------------------------------------------------------------------------
__global__ __launch_bounds__(256)
void qkv_proj_kernel(const float* __restrict__ x,
                     const float* __restrict__ Wq, const float* __restrict__ bq,
                     const float* __restrict__ Wk, const float* __restrict__ bk,
                     const float* __restrict__ Wv, const float* __restrict__ bvec,
                     __bf16* __restrict__ qb, __bf16* __restrict__ kb,
                     __bf16* __restrict__ vb) {
  const int w    = blockIdx.x * 8 + (threadIdx.x >> 5);
  const int lane = threadIdx.x & 31;
  const int r    = lane & 15;   // A row / B col / D col within tile
  const int h    = lane >> 4;   // lane half

  const int b   = w / (20 * 256);
  const int rem = w % (20 * 256);
  const int o0  = (rem >> 8) * 16;   // output-row tile base (0..304)
  const int n0  = (rem & 255) * 16;  // pixel tile base

  const float* Wm; const float* bias; __bf16* dst; int orow;
  if (o0 < 32)      { Wm = Wq; bias = bq;   dst = qb + (size_t)b * DQK_ * N_; orow = o0;      }
  else if (o0 < 64) { Wm = Wk; bias = bk;   dst = kb + (size_t)b * DQK_ * N_; orow = o0 - 32; }
  else              { Wm = Wv; bias = bvec; dst = vb + (size_t)b * C_  * N_;  orow = o0 - 64; }

  f32x8 acc = {};
  for (int c0 = 0; c0 < C_; c0 += 32) {
    bf16x16 av, bm;
    #pragma unroll
    for (int j = 0; j < 16; ++j)
      av[j] = (__bf16)Wm[(orow + r) * C_ + c0 + a_kk(j, h)];
    #pragma unroll
    for (int j = 0; j < 16; ++j)   // B[kk][n] = x[b, c0+kk, n0+n], kk = 16h+j
      bm[j] = (__bf16)x[(size_t)(b * C_ + c0 + 16 * h + j) * N_ + n0 + r];
    acc = wmma_bf16(av, bm, acc);
  }
  #pragma unroll
  for (int i = 0; i < 8; ++i) {
    const int m = orow + i + 8 * h;
    dst[(size_t)m * N_ + n0 + r] = (__bf16)(acc[i] + bias[m]);
  }
}

// ---------------------------------------------------------------------------
// Kernel 2: energy[b,n,m] = sum_d q[b,d,n]*k[b,d,m].  K-dim = 32 exactly, so
// each 16x16 tile is ONE wmma.  A wave keeps one Q A-tile and sweeps 8
// consecutive m-tiles (A gathered once).  Energy is streamed out with NT
// stores (256 MB write-once, > L2).
// ---------------------------------------------------------------------------
__global__ __launch_bounds__(256)
void energy_kernel(const __bf16* __restrict__ qb, const __bf16* __restrict__ kb,
                   float* __restrict__ attn) {
  const int w    = blockIdx.x * 8 + (threadIdx.x >> 5);
  const int lane = threadIdx.x & 31;
  const int r    = lane & 15;
  const int h    = lane >> 4;

  const int b   = w >> 13;            // 8192 waves per batch
  const int rem = w & 8191;
  const int n0  = (rem >> 5) * 16;    // 256 n-tiles
  const int m0  = (rem & 31) * 128;   // 32 groups of 8 m-tiles

  const __bf16* qbb = qb + (size_t)b * DQK_ * N_;
  const __bf16* kbb = kb + (size_t)b * DQK_ * N_;

  bf16x16 av;                         // A[n_row][d] = q[b, d, n0+n_row]
  #pragma unroll
  for (int j = 0; j < 16; ++j)
    av[j] = qbb[(size_t)a_kk(j, h) * N_ + n0 + r];

  float* out = attn + (size_t)b * N_ * N_;
  for (int mt = 0; mt < 8; ++mt) {
    const int mb = m0 + mt * 16;
    bf16x16 bm;                       // B[d][m_col] = k[b, d, mb+m_col]
    #pragma unroll
    for (int j = 0; j < 16; ++j)
      bm[j] = kbb[(size_t)(16 * h + j) * N_ + mb + r];
    f32x8 acc = {};
    acc = wmma_bf16(av, bm, acc);
    #pragma unroll
    for (int i = 0; i < 8; ++i)
      __builtin_nontemporal_store(acc[i],
          &out[(size_t)(n0 + i + 8 * h) * N_ + mb + r]);
  }
}

// ---------------------------------------------------------------------------
// Kernel 3: in-place row softmax over m.  One block per (b,n) row; each
// thread owns a contiguous 64 B slice staged global->LDS with the CDNA5
// async path (ASYNCcnt) when the toolchain exposes it, then kept in regs.
// ---------------------------------------------------------------------------
__global__ __launch_bounds__(256)
void softmax_kernel(float* __restrict__ attn) {
#if HAVE_ASYNC_LDS
  __shared__ float buf[N_];
#endif
  __shared__ float redmax[8];
  __shared__ float redsum[8];

  float* row     = attn + (size_t)blockIdx.x * N_;
  const int t    = threadIdx.x;
  const int base = t * 16;          // 16 contiguous floats per thread

  float v[16];
#if HAVE_ASYNC_LDS
  {
    gv4i* g = (gv4i*)(row + base);
    sv4i* l = (sv4i*)(buf + base);
    __builtin_amdgcn_global_load_async_to_lds_b128(g, l, 0,  0);
    __builtin_amdgcn_global_load_async_to_lds_b128(g, l, 16, 0);
    __builtin_amdgcn_global_load_async_to_lds_b128(g, l, 32, 0);
    __builtin_amdgcn_global_load_async_to_lds_b128(g, l, 48, 0);
    __builtin_amdgcn_s_wait_asynccnt(0);   // own slice complete -> safe to read
    #pragma unroll
    for (int j = 0; j < 16; ++j) v[j] = buf[base + j];
  }
#else
  {
    const f32x4* p = (const f32x4*)(row + base);
    #pragma unroll
    for (int q = 0; q < 4; ++q) {
      const f32x4 f = p[q];
      v[4*q+0] = f[0]; v[4*q+1] = f[1]; v[4*q+2] = f[2]; v[4*q+3] = f[3];
    }
  }
#endif

  float lmax = v[0];
  #pragma unroll
  for (int j = 1; j < 16; ++j) lmax = fmaxf(lmax, v[j]);
  #pragma unroll
  for (int off = 16; off > 0; off >>= 1)
    lmax = fmaxf(lmax, __shfl_down(lmax, off, 32));
  if ((t & 31) == 0) redmax[t >> 5] = lmax;
  __syncthreads();
  float rmax = redmax[0];
  #pragma unroll
  for (int i = 1; i < 8; ++i) rmax = fmaxf(rmax, redmax[i]);

  float lsum = 0.0f;
  #pragma unroll
  for (int j = 0; j < 16; ++j) { v[j] = __expf(v[j] - rmax); lsum += v[j]; }
  #pragma unroll
  for (int off = 16; off > 0; off >>= 1)
    lsum += __shfl_down(lsum, off, 32);
  if ((t & 31) == 0) redsum[t >> 5] = lsum;
  __syncthreads();
  float s = 0.0f;
  #pragma unroll
  for (int i = 0; i < 8; ++i) s += redsum[i];

  const float inv = 1.0f / s;
  f32x4* op = (f32x4*)(row + base);
  #pragma unroll
  for (int q = 0; q < 4; ++q) {
    f32x4 f;
    f[0] = v[4*q+0] * inv; f[1] = v[4*q+1] * inv;
    f[2] = v[4*q+2] * inv; f[3] = v[4*q+3] * inv;
    __builtin_nontemporal_store(f, op + q);   // streaming write-back
  }
}

// ---------------------------------------------------------------------------
// Kernel 4: out[b,c,n] = sum_m v[b,c,m]*attn[b,n,m]; fused epilogue
// weighted = gamma*out, final = weighted + x.  B-matrix columns are
// contiguous attention rows -> NT b128 loads + on-the-fly fp32->bf16.
// ---------------------------------------------------------------------------
__global__ __launch_bounds__(256)
void attn_v_kernel(const __bf16* __restrict__ vb, const float* __restrict__ attn,
                   const float* __restrict__ x, const float* __restrict__ gamma,
                   float* __restrict__ final_, float* __restrict__ wout) {
  const int w    = blockIdx.x * 8 + (threadIdx.x >> 5);
  const int lane = threadIdx.x & 31;
  const int r    = lane & 15;
  const int h    = lane >> 4;

  const int b   = w >> 12;           // / 4096 tiles per batch
  const int rem = w & 4095;
  const int c0  = (rem >> 8) * 16;
  const int n0  = (rem & 255) * 16;

  const __bf16* vrow = vb + ((size_t)b * C_ + c0 + r) * N_;            // A rows
  const float*  arow = attn + ((size_t)b * N_ + n0 + r) * N_ + 16 * h; // B cols

  f32x8 acc = {};
  for (int m0 = 0; m0 < N_; m0 += 32) {
    bf16x16 av, bm;
    #pragma unroll
    for (int j = 0; j < 16; ++j)
      av[j] = vrow[m0 + a_kk(j, h)];

    const f32x4* p = (const f32x4*)(arow + m0);   // 64B-aligned
    const f32x4 q0 = __builtin_nontemporal_load(p + 0);
    const f32x4 q1 = __builtin_nontemporal_load(p + 1);
    const f32x4 q2 = __builtin_nontemporal_load(p + 2);
    const f32x4 q3 = __builtin_nontemporal_load(p + 3);
    #pragma unroll
    for (int j = 0; j < 4; ++j) {
      bm[j]      = (__bf16)q0[j];
      bm[j + 4]  = (__bf16)q1[j];
      bm[j + 8]  = (__bf16)q2[j];
      bm[j + 12] = (__bf16)q3[j];
    }

    if (m0 + 512 < N_)               // stream-prefetch upcoming attention data
      __builtin_prefetch(arow + m0 + 512, 0, 0);

    acc = wmma_bf16(av, bm, acc);
  }

  const float g = gamma[0];
  #pragma unroll
  for (int i = 0; i < 8; ++i) {
    const int c = c0 + i + 8 * h;
    const size_t idx = ((size_t)b * C_ + c) * N_ + n0 + r;
    const float wv = g * acc[i];
    __builtin_nontemporal_store(wv, &wout[idx]);
    __builtin_nontemporal_store(wv + x[idx], &final_[idx]);
  }
}

// ---------------------------------------------------------------------------
extern "C" void kernel_launch(void* const* d_in, const int* in_sizes, int n_in,
                              void* d_out, int out_size, void* d_ws, size_t ws_size,
                              hipStream_t stream) {
  const float* x     = (const float*)d_in[0];
  const float* Wq    = (const float*)d_in[1];
  const float* bq    = (const float*)d_in[2];
  const float* Wk    = (const float*)d_in[3];
  const float* bk    = (const float*)d_in[4];
  const float* Wv    = (const float*)d_in[5];
  const float* bvv   = (const float*)d_in[6];
  const float* gamma = (const float*)d_in[7];

  float* final_ = (float*)d_out;                        // [4,256,4096]
  float* wout   = final_ + (size_t)B_ * C_ * N_;        // [4,256,4096]
  float* attn   = wout   + (size_t)B_ * C_ * N_;        // [4,4096,4096]

  // Workspace: bf16 q/k/v (~10.5 MB total)
  __bf16* qb = (__bf16*)d_ws;
  __bf16* kb = qb + (size_t)B_ * DQK_ * N_;
  __bf16* vb = kb + (size_t)B_ * DQK_ * N_;

  // 1) QKV projection: 4*20*256 tiles / 8 waves = 2560 blocks
  qkv_proj_kernel<<<2560, 256, 0, stream>>>(x, Wq, bq, Wk, bk, Wv, bvv, qb, kb, vb);
  // 2) energy: 4*256*32 wave-strips / 8 = 4096 blocks (8 m-tiles per wave)
  energy_kernel<<<4096, 256, 0, stream>>>(qb, kb, attn);
  // 3) softmax: one block per row
  softmax_kernel<<<B_ * N_, 256, 0, stream>>>(attn);
  // 4) V x attn^T + epilogue: 4*16*256 tiles / 8 = 2048 blocks
  attn_v_kernel<<<2048, 256, 0, stream>>>(vb, attn, x, gamma, final_, wout);
}
